// TharvexalFuelLayer_54795192762560
// MI455X (gfx1250) — compile-verified, matrix operand order checked by reading, not run on hardware
//
#include <hip/hip_runtime.h>
#include <math.h>
#include <stdint.h>

// ---- problem constants ----
#define BB 2
#define SS 2048
#define DD 512
#define HH 1024
#define EE 8
#define QQ 4
#define NTOK (BB*SS)        // 4096
#define ND   (NTOK*DD)      // 2097152

// ---- workspace layout ----
// float region:
#define WS_MIX   0
#define WS_C     (ND)                        // NTOK*EE*QQ
#define WS_GATE  (WS_C + NTOK*EE*QQ)
#define WS_USAGE (WS_GATE + NTOK)
#define WS_CONF  (WS_USAGE + EE)
#define WS_TOTAL (WS_CONF + 1)
// bf16 region (pre-converted transposed expert weights) starts at:
#define WS_BF_BYTE_OFF ((((size_t)WS_TOTAL * 4) + 255) & ~(size_t)255)
#define W_ELEMS ((size_t)EE * QQ * DD * HH)  // 16,777,216 per weight tensor

typedef __attribute__((ext_vector_type(16))) __bf16 v16bf;
typedef __attribute__((ext_vector_type(8)))  __bf16 v8bf;
typedef __attribute__((ext_vector_type(8)))  float  v8f;
typedef int v4i __attribute__((__vector_size__(16)));   // matches builtin param

#define AS1 __attribute__((address_space(1)))
#define AS3 __attribute__((address_space(3)))

#if __has_builtin(__builtin_amdgcn_global_load_async_to_lds_b128) && \
    __has_builtin(__builtin_amdgcn_s_wait_asynccnt)
#define USE_ASYNC 1
#else
#define USE_ASYNC 0
#endif

#if USE_ASYNC
#define ASYNC_WAIT() __builtin_amdgcn_s_wait_asynccnt(0)
#else
#define ASYNC_WAIT() ((void)0)
#endif

__device__ __forceinline__ float gelu_exact(float v) {
    return 0.5f * v * (1.0f + erff(v * 0.70710678118654752f));
}
__device__ __forceinline__ float sigmoidf(float v) { return 1.0f / (1.0f + expf(-v)); }

__device__ __forceinline__ v8f zero8() {
    v8f r;
#pragma unroll
    for (int i = 0; i < 8; i++) r[i] = 0.0f;
    return r;
}

// Load a 16x32 bf16 WMMA operand fragment from row-major LDS (rows x K).
// Lane layout (ISA 7.12.2): lanes 0-15 row=lane, K {k0..k0+7, k0+16..k0+23};
// lanes 16-31 row=lane-16, K shifted by +8.
__device__ __forceinline__ v16bf ld_frag16(const __bf16* base, int stride, int k0,
                                           int hi, int lm) {
    const __bf16* p = base + lm * stride + k0 + hi * 8;
    v8bf lo = *(const v8bf*)(p);
    v8bf hs = *(const v8bf*)(p + 16);
    v16bf r;
#pragma unroll
    for (int i = 0; i < 8; i++) { r[i] = lo[i]; r[8 + i] = hs[i]; }
    return r;
}

#define WMMA_BF16(A, B, C) \
    __builtin_amdgcn_wmma_f32_16x16x32_bf16(false, (A), false, (B), (short)0, (C), false, false)

// ---------------------------------------------------------------------------
__global__ void k_zero(float* ws, int n) {
    int i = blockIdx.x * blockDim.x + threadIdx.x;
    if (i < n) ws[i] = 0.0f;
}

// ---------------------------------------------------------------------------
// Tiled transpose + f32->bf16 convert: src (K,N) f32 -> dst (N,K) bf16.
// grid = (N/32, K/32, EE*QQ), block = 256 (32x8).
__global__ __launch_bounds__(256) void k_xpose(const float* __restrict__ src,
                                               __bf16* __restrict__ dst,
                                               int K, int N) {
    __shared__ float tile[32][33];
    const size_t sb = (size_t)blockIdx.z * K * N;
    const int n0 = blockIdx.x * 32, k0 = blockIdx.y * 32;
    const int tx = threadIdx.x & 31, ty = threadIdx.x >> 5;
#pragma unroll
    for (int i = 0; i < 32; i += 8)
        tile[ty + i][tx] = src[sb + (size_t)(k0 + ty + i) * N + (n0 + tx)];
    __syncthreads();
#pragma unroll
    for (int i = 0; i < 32; i += 8)
        dst[sb + (size_t)(n0 + ty + i) * K + (k0 + tx)] = (__bf16)tile[tx][ty + i];
}

// ---------------------------------------------------------------------------
__device__ __forceinline__ float block_sum(float v, float* red) {
    int t = threadIdx.x;
    red[t] = v;
    __syncthreads();
    for (int s = 128; s > 0; s >>= 1) {
        if (t < s) red[t] += red[t + s];
        __syncthreads();
    }
    float r = red[0];
    __syncthreads();
    return r;
}

__global__ __launch_bounds__(256) void k_router(
    const float* __restrict__ x,
    const float* __restrict__ r0w1, const float* __restrict__ r0b1,
    const float* __restrict__ r0g,  const float* __restrict__ r0be,
    const float* __restrict__ r0w2, const float* __restrict__ r0b2,
    const float* __restrict__ r1w1, const float* __restrict__ r1b1,
    const float* __restrict__ r1g,  const float* __restrict__ r1be,
    const float* __restrict__ r1w2, const float* __restrict__ r1b2,
    const float* __restrict__ r2w1, const float* __restrict__ r2b1,
    const float* __restrict__ r2g,  const float* __restrict__ r2be,
    const float* __restrict__ r2w2, const float* __restrict__ r2b2,
    const float* __restrict__ cw1,  const float* __restrict__ cb1,
    const float* __restrict__ cw2,  const float* __restrict__ cb2,
    const float* __restrict__ ent,  const float* __restrict__ gw,
    const float* __restrict__ gb,   const float* __restrict__ rgw,
    const float* __restrict__ rgb,
    float* __restrict__ ws, float* __restrict__ rw_out) {
    __shared__ float xs[DD];
    __shared__ float hs[DD / 2];
    __shared__ float red[256];
    __shared__ float logits[EE];
    __shared__ float rw[EE];
    __shared__ float spr[EE][QQ];
    __shared__ float epb[EE][QQ];

    const int tok = blockIdx.x;
    const int t = threadIdx.x;
    const float* xp = x + (size_t)tok * DD;
    xs[t] = xp[t];
    xs[t + 256] = xp[t + 256];
    if (t < EE) logits[t] = 0.0f;
    __syncthreads();

    const float* W1s[3] = {r0w1, r1w1, r2w1};
    const float* B1s[3] = {r0b1, r1b1, r2b1};
    const float* Gs[3]  = {r0g,  r1g,  r2g};
    const float* BEs[3] = {r0be, r1be, r2be};
    const float* W2s[3] = {r0w2, r1w2, r2w2};
    const float* B2s[3] = {r0b2, r1b2, r2b2};
    const int OUTS[3] = {8, 4, 2};

    for (int lvl = 0; lvl < 3; lvl++) {
        const float* w1 = W1s[lvl];
        float acc = B1s[lvl][t];
        for (int d = 0; d < DD; d++) acc += xs[d] * w1[d * 256 + t];
        float h = gelu_exact(acc);
        float m = block_sum(h, red) * (1.0f / 256.0f);
        float dv = h - m;
        float var = block_sum(dv * dv, red) * (1.0f / 256.0f);
        float hn = dv * rsqrtf(var + 1e-5f) * Gs[lvl][t] + BEs[lvl][t];
        hs[t] = hn;
        __syncthreads();
        int nout = OUTS[lvl];
        if (t < nout) {
            const float* w2 = W2s[lvl];
            float lg = B2s[lvl][t];
            for (int k = 0; k < 256; k++) lg += hs[k] * w2[k * nout + t];
            if (lvl == 0) logits[t] += lg;
            else if (lvl == 1) { logits[t] += lg; logits[t + 4] += lg; }
            else { logits[t] += lg; logits[t + 2] += lg; logits[t + 4] += lg; logits[t + 6] += lg; }
        }
        __syncthreads();
    }

    // confidence scorer (feeds aux only)
    if (t < 64) {
        float a = cb1[t];
        for (int d = 0; d < DD; d++) a += xs[d] * cw1[d * 64 + t];
        red[t] = tanhf(a);
    }
    __syncthreads();
    if (t == 0) {
        float c = cb2[0];
        for (int j = 0; j < 64; j++) c += red[j] * cw2[j];
        c = sigmoidf(c);
        atomicAdd(&ws[WS_CONF], fabsf(c - 0.7f));
    }
    __syncthreads();

    // softmax + top-2 + renorm
    if (t == 0) {
        float mx = -1e30f;
        for (int j = 0; j < EE; j++) mx = fmaxf(mx, logits[j]);
        float p[EE], s = 0.0f;
        for (int j = 0; j < EE; j++) { p[j] = expf(logits[j] - mx); s += p[j]; }
        for (int j = 0; j < EE; j++) p[j] /= s;
        int i1 = 0;
        for (int j = 1; j < EE; j++) if (p[j] > p[i1]) i1 = j;
        int i2 = -1;
        for (int j = 0; j < EE; j++) if (j != i1 && (i2 < 0 || p[j] > p[i2])) i2 = j;
        float den = p[i1] + p[i2] + 1e-10f;
        for (int j = 0; j < EE; j++) rw[j] = (j == i1 || j == i2) ? p[j] / den : 0.0f;
    }
    __syncthreads();
    if (t < EE) {
        rw_out[tok * EE + t] = rw[t];
        atomicAdd(&ws[WS_USAGE + t], rw[t]);
    }

    // state probs + entanglement -> per-(e,q) coefficient
    if (t < EE * QQ) {
        int e = t >> 2, q = t & 3;
        float a = gb[e * QQ + q];
        const float* gp = gw + (size_t)e * DD * QQ + q;
        for (int d = 0; d < DD; d++) a += xs[d] * gp[d * QQ];
        spr[e][q] = a;
    }
    __syncthreads();
    if (t < EE) {
        int e = t;
        float mx = -1e30f;
        for (int q = 0; q < QQ; q++) mx = fmaxf(mx, spr[e][q]);
        float sp[QQ], s = 0.0f;
        for (int q = 0; q < QQ; q++) { sp[q] = expf(spr[e][q] - mx); s += sp[q]; }
        for (int q = 0; q < QQ; q++) sp[q] /= s;
        float er[QQ];
        float mx2 = -1e30f;
        for (int r = 0; r < QQ; r++) {
            float a = 0.0f;
            for (int q = 0; q < QQ; q++) a += sp[q] * ent[(e * QQ + q) * QQ + r];
            er[r] = a;
            mx2 = fmaxf(mx2, a);
        }
        float s2 = 0.0f;
        for (int r = 0; r < QQ; r++) { er[r] = expf(er[r] - mx2); s2 += er[r]; }
        for (int r = 0; r < QQ; r++) epb[e][r] = er[r] / s2;
    }
    __syncthreads();
    if (t < EE * QQ) {
        int e = t >> 2, q = t & 3;
        ws[WS_C + ((size_t)tok * EE + e) * QQ + q] = rw[e] * epb[e][q];
    }

    // residual gate
    float gp = xs[t] * rgw[t] + xs[t + 256] * rgw[t + 256];
    float gs = block_sum(gp, red);
    if (t == 0) ws[WS_GATE + tok] = sigmoidf(gs + rgb[0]);
}

// ---------------------------------------------------------------------------
// Expert stack: per (64-token tile, expert e), loop q: GEMM1(64x512x1024) ->
// gelu -> LDS h -> GEMM2(64x1024x512) -> atomic accumulate c*so into mix.
// B tiles come from pre-transposed bf16 weights via async LDS copies
// (double-buffered, ASYNCcnt-tracked) when the toolchain exposes them.
#define KA 520
#define KH 1032
#define KB 40
#define LDS_AS_BYTES (64 * KA * 2)          // 66560
#define LDS_HS_BYTES (64 * KH * 2)          // 132096
#define LDS_BS_BYTES (128 * KB * 2)         // 10240 per buffer
#define LDS_TOTAL (LDS_AS_BYTES + LDS_HS_BYTES + 2 * LDS_BS_BYTES + 64 * 4)

// Stage one 128(n) x 32(k) bf16 tile: Wt row-major (N,Kdim), rows contiguous
// in k -> 16B chunks map 1:1 onto LDS rows of stride KB.
__device__ __forceinline__ void stage_tileB(const __bf16* __restrict__ Wt, int Kdim,
                                            int nbase, int kk, __bf16* Bs) {
    const int t = threadIdx.x;
#pragma unroll
    for (int rep = 0; rep < 2; rep++) {
        int c = t + rep * 256;               // 0..511 chunk id
        int n = c >> 2, kc = (c & 3) * 8;
        const __bf16* g = Wt + (size_t)(nbase + n) * Kdim + (kk + kc);
        __bf16* l = Bs + n * KB + kc;
#if USE_ASYNC
        __builtin_amdgcn_global_load_async_to_lds_b128(
            (AS1 v4i*)(uintptr_t)g, (AS3 v4i*)(uint32_t)(uintptr_t)l, 0, 0);
#else
        *(v8bf*)l = *(const v8bf*)g;
#endif
    }
}

__global__ __launch_bounds__(256) void k_expert(
    const float* __restrict__ x, const float* __restrict__ semb,
    const __bf16* __restrict__ w1t,   // (E,Q,H,D) bf16: GEMM1 B transposed
    const float* __restrict__ eb1,
    const __bf16* __restrict__ w2t,   // (E,Q,D,H) bf16: GEMM2 B transposed
    const float* __restrict__ eb2,
    const float* __restrict__ ws_c, float* __restrict__ mix) {
    extern __shared__ char smem[];
    __bf16* As  = (__bf16*)smem;
    __bf16* Hs  = (__bf16*)(smem + LDS_AS_BYTES);
    __bf16* Bs0 = (__bf16*)(smem + LDS_AS_BYTES + LDS_HS_BYTES);
    __bf16* Bs1 = (__bf16*)(smem + LDS_AS_BYTES + LDS_HS_BYTES + LDS_BS_BYTES);
    float*  cb  = (float*)(smem + LDS_AS_BYTES + LDS_HS_BYTES + 2 * LDS_BS_BYTES);
    __bf16* Bsb[2] = {Bs0, Bs1};

    const int t = threadIdx.x;
    const int lane = t & 31, wave = t >> 5;
    const int wm = wave >> 2, wn = wave & 3;   // 2 x 4 wave grid
    const int hi = lane >> 4, lm = lane & 15;
    const int tok0 = blockIdx.x * 64;
    const int e = blockIdx.y;

    for (int q = 0; q < QQ; q++) {
        // ---- stage A tile: bf16(x + state_emb[e][q]) ----
        const float* sep = semb + (size_t)(e * QQ + q) * DD;
        for (int i = 0; i < 128; i++) {
            int idx = t + i * 256;
            int m = idx >> 9, k = idx & 511;
            As[m * KA + k] = (__bf16)(x[(size_t)(tok0 + m) * DD + k] + sep[k]);
        }
        if (t < 64) cb[t] = ws_c[((size_t)(tok0 + t) * EE + e) * QQ + q];
        __syncthreads();

        // ---- GEMM1: Hs = gelu(As @ ew1[e,q] + eb1[e,q]) ----
        const __bf16* W1T = w1t + (size_t)(e * QQ + q) * HH * DD;  // (H rows, D cols)
        const float*  B1  = eb1 + (size_t)(e * QQ + q) * HH;
        for (int nb = 0; nb < HH / 128; nb++) {
            v8f acc[2][2];
#pragma unroll
            for (int a = 0; a < 2; a++)
#pragma unroll
                for (int b = 0; b < 2; b++) acc[a][b] = zero8();

            stage_tileB(W1T, DD, nb * 128, 0, Bsb[0]);
            ASYNC_WAIT();
            __syncthreads();
            for (int kk = 0; kk < DD; kk += 32) {
                const int cur = (kk >> 5) & 1;
                if (kk + 32 < DD) stage_tileB(W1T, DD, nb * 128, kk + 32, Bsb[1 - cur]);
                const __bf16* B = Bsb[cur];
                v16bf a0 = ld_frag16(As + (wm * 32) * KA, KA, kk, hi, lm);
                v16bf a1 = ld_frag16(As + (wm * 32 + 16) * KA, KA, kk, hi, lm);
                v16bf b0 = ld_frag16(B + (wn * 32) * KB, KB, 0, hi, lm);
                v16bf b1 = ld_frag16(B + (wn * 32 + 16) * KB, KB, 0, hi, lm);
                acc[0][0] = WMMA_BF16(a0, b0, acc[0][0]);
                acc[0][1] = WMMA_BF16(a0, b1, acc[0][1]);
                acc[1][0] = WMMA_BF16(a1, b0, acc[1][0]);
                acc[1][1] = WMMA_BF16(a1, b1, acc[1][1]);
                ASYNC_WAIT();
                __syncthreads();
            }
            // C layout: VGPR i -> M = i + 8*hi, N = lm (ISA 7.12.2)
#pragma unroll
            for (int fm = 0; fm < 2; fm++)
#pragma unroll
                for (int fn = 0; fn < 2; fn++)
#pragma unroll
                    for (int i = 0; i < 8; i++) {
                        int row = wm * 32 + fm * 16 + hi * 8 + i;
                        int col = nb * 128 + wn * 32 + fn * 16 + lm;
                        float v = acc[fm][fn][i] + B1[col];
                        Hs[row * KH + col] = (__bf16)gelu_exact(v);
                    }
        }
        __syncthreads();

        // ---- GEMM2: so = Hs @ ew2[e,q] + eb2[e,q]; mix += c * so ----
        const __bf16* W2T = w2t + (size_t)(e * QQ + q) * DD * HH;  // (D rows, H cols)
        const float*  B2  = eb2 + (size_t)(e * QQ + q) * DD;
        for (int nb = 0; nb < DD / 128; nb++) {
            v8f acc[2][2];
#pragma unroll
            for (int a = 0; a < 2; a++)
#pragma unroll
                for (int b = 0; b < 2; b++) acc[a][b] = zero8();

            stage_tileB(W2T, HH, nb * 128, 0, Bsb[0]);
            ASYNC_WAIT();
            __syncthreads();
            for (int kk = 0; kk < HH; kk += 32) {
                const int cur = (kk >> 5) & 1;
                if (kk + 32 < HH) stage_tileB(W2T, HH, nb * 128, kk + 32, Bsb[1 - cur]);
                const __bf16* B = Bsb[cur];
                v16bf a0 = ld_frag16(Hs + (wm * 32) * KH, KH, kk, hi, lm);
                v16bf a1 = ld_frag16(Hs + (wm * 32 + 16) * KH, KH, kk, hi, lm);
                v16bf b0 = ld_frag16(B + (wn * 32) * KB, KB, 0, hi, lm);
                v16bf b1 = ld_frag16(B + (wn * 32 + 16) * KB, KB, 0, hi, lm);
                acc[0][0] = WMMA_BF16(a0, b0, acc[0][0]);
                acc[0][1] = WMMA_BF16(a0, b1, acc[0][1]);
                acc[1][0] = WMMA_BF16(a1, b0, acc[1][0]);
                acc[1][1] = WMMA_BF16(a1, b1, acc[1][1]);
                ASYNC_WAIT();
                __syncthreads();
            }
#pragma unroll
            for (int fm = 0; fm < 2; fm++)
#pragma unroll
                for (int fn = 0; fn < 2; fn++)
#pragma unroll
                    for (int i = 0; i < 8; i++) {
                        int row = wm * 32 + fm * 16 + hi * 8 + i;
                        int col = nb * 128 + wn * 32 + fn * 16 + lm;
                        float v = (acc[fm][fn][i] + B2[col]) * cb[row];
                        atomicAdd(&mix[(size_t)(tok0 + row) * DD + col], v);
                    }
        }
        __syncthreads();  // before next q overwrites As/Hs
    }
}

// ---------------------------------------------------------------------------
__global__ __launch_bounds__(256) void k_combine(
    const float* __restrict__ x, const float* __restrict__ mix,
    const float* __restrict__ gate, const float* __restrict__ lng,
    const float* __restrict__ lnb, float* __restrict__ out) {
    __shared__ float red[256];
    const int tok = blockIdx.x, t = threadIdx.x;
    float g = gate[tok];
    size_t base = (size_t)tok * DD;
    float y0 = g * mix[base + t] + (1.0f - g) * x[base + t];
    float y1 = g * mix[base + t + 256] + (1.0f - g) * x[base + t + 256];
    float m = block_sum(y0 + y1, red) * (1.0f / 512.0f);
    float d0 = y0 - m, d1 = y1 - m;
    float var = block_sum(d0 * d0 + d1 * d1, red) * (1.0f / 512.0f);
    float r = rsqrtf(var + 1e-5f);
    out[base + t]       = d0 * r * lng[t] + lnb[t];
    out[base + t + 256] = d1 * r * lng[t + 256] + lnb[t + 256];
}

__global__ void k_aux(const float* __restrict__ ws, float* __restrict__ aux) {
    float tot = 0.0f;
    for (int j = 0; j < EE; j++) tot += ws[WS_USAGE + j];
    float mean = tot / EE;
    float a = 0.0f;
    for (int j = 0; j < EE; j++) {
        float d = ws[WS_USAGE + j] - mean;
        a += d * d;
    }
    a /= EE;
    a += 0.1f * ws[WS_CONF] / (float)NTOK;
    aux[0] = a;
}

// ---------------------------------------------------------------------------
extern "C" void kernel_launch(void* const* d_in, const int* in_sizes, int n_in,
                              void* d_out, int out_size, void* d_ws, size_t ws_size,
                              hipStream_t stream) {
    const float* x    = (const float*)d_in[0];
    const float* r0w1 = (const float*)d_in[1];
    const float* r0b1 = (const float*)d_in[2];
    const float* r0g  = (const float*)d_in[3];
    const float* r0be = (const float*)d_in[4];
    const float* r0w2 = (const float*)d_in[5];
    const float* r0b2 = (const float*)d_in[6];
    const float* r1w1 = (const float*)d_in[7];
    const float* r1b1 = (const float*)d_in[8];
    const float* r1g  = (const float*)d_in[9];
    const float* r1be = (const float*)d_in[10];
    const float* r1w2 = (const float*)d_in[11];
    const float* r1b2 = (const float*)d_in[12];
    const float* r2w1 = (const float*)d_in[13];
    const float* r2b1 = (const float*)d_in[14];
    const float* r2g  = (const float*)d_in[15];
    const float* r2be = (const float*)d_in[16];
    const float* r2w2 = (const float*)d_in[17];
    const float* r2b2 = (const float*)d_in[18];
    const float* cw1  = (const float*)d_in[19];
    const float* cb1  = (const float*)d_in[20];
    const float* cw2  = (const float*)d_in[21];
    const float* cb2  = (const float*)d_in[22];
    const float* semb = (const float*)d_in[23];
    const float* ew1  = (const float*)d_in[24];
    const float* eb1  = (const float*)d_in[25];
    const float* ew2  = (const float*)d_in[26];
    const float* eb2  = (const float*)d_in[27];
    const float* ent  = (const float*)d_in[28];
    const float* gw   = (const float*)d_in[29];
    const float* gb   = (const float*)d_in[30];
    const float* rgw  = (const float*)d_in[31];
    const float* rgb  = (const float*)d_in[32];
    const float* lng  = (const float*)d_in[33];
    const float* lnb  = (const float*)d_in[34];

    float*  ws  = (float*)d_ws;
    __bf16* w1t = (__bf16*)((char*)d_ws + WS_BF_BYTE_OFF);
    __bf16* w2t = w1t + W_ELEMS;
    float*  out = (float*)d_out;           // (B,S,D) flat
    float*  aux = out + ND;                // scalar
    float*  rwo = out + ND + 1;            // (B,S,E) flat

    k_zero<<<(WS_TOTAL + 255) / 256, 256, 0, stream>>>(ws, WS_TOTAL);

    // pre-convert expert weights to bf16, transposed to (n, k) row-major
    {
        dim3 g1(HH / 32, DD / 32, EE * QQ);   // ew1 (D,H) -> w1t (H,D)
        k_xpose<<<g1, 256, 0, stream>>>(ew1, w1t, DD, HH);
        dim3 g2(DD / 32, HH / 32, EE * QQ);   // ew2 (H,D) -> w2t (D,H)
        k_xpose<<<g2, 256, 0, stream>>>(ew2, w2t, HH, DD);
    }

    k_router<<<NTOK, 256, 0, stream>>>(
        x, r0w1, r0b1, r0g, r0be, r0w2, r0b2,
        r1w1, r1b1, r1g, r1be, r1w2, r1b2,
        r2w1, r2b1, r2g, r2be, r2w2, r2b2,
        cw1, cb1, cw2, cb2, ent, gw, gb, rgw, rgb,
        ws, rwo);

    dim3 grid(NTOK / 64, EE);
    k_expert<<<grid, 256, LDS_TOTAL, stream>>>(
        x, semb, w1t, eb1, w2t, eb2, ws + WS_C, ws + WS_MIX);

    k_combine<<<NTOK, 256, 0, stream>>>(x, ws + WS_MIX, ws + WS_GATE, lng, lnb, out);

    k_aux<<<1, 1, 0, stream>>>(ws, aux);
}
